// Attention_28063316312444
// MI455X (gfx1250) — matrix operand order repaired and checked by reading, not compiled, not run on hardware
//
#include <hip/hip_runtime.h>
#include <hip/hip_bf16.h>
#include <math.h>

// ---------------------------------------------------------------------------
// Attention kernel for MI455X (gfx1250, wave32, WMMA).
//   q = mlp(x), k = mlp(x)  (129 -> 64 -> 64, ReLU)
//   S = softmax_row( 10 * (q k^T) / rowmax[col] )   (torch broadcast quirk)
//
// Roofline: output is 8192^2 fp32 = 256 MB -> ~11us floor @ 23.3 TB/s HBM.
// qk^T is only 8.6 GFLOP; it is recomputed in 3 WMMA sweeps instead of ever
// storing the 256 MB logit matrix.  q/k are 1 MB f16 each and live in L2
// (192 MB) between passes; register-blocking (4 row tiles per wave) keeps
// the L2 re-read traffic of the sweeps ~128 MB/pass.  Sweep loops are
// unrolled 2x so two independent WMMA pipelines cover the WMMA->VALU
// hazard slots.
// ---------------------------------------------------------------------------

typedef __attribute__((ext_vector_type(16))) _Float16 v16h;
typedef __attribute__((ext_vector_type(8)))  _Float16 v8h;
typedef __attribute__((ext_vector_type(8)))  float    v8f;

constexpr int N    = 8192;
constexpr int FIN  = 129;
constexpr int D    = 64;
constexpr int NT   = N / 16;      // 512 column/row tiles
constexpr int RT   = 4;           // row tiles per wave in the sweep passes
constexpr float TEMP = 10.0f;
constexpr float NEG_BIG = -3.0e38f;

// ---- WMMA fragment loaders (contiguous k order, same in A and B, wave32) ----------

// A: 16x32 f16 (M x K). Lane L (0-15): M=L, K=0..15 contiguous;
// lanes 16-31: same M, K=16..31 contiguous.  One 32-byte chunk per lane.
__device__ __forceinline__ v16h load_a_frag(const _Float16* __restrict__ base,
                                            int rowBase, int kOff, int lane) {
  const int m  = lane & 15;
  const int kb = kOff + ((lane & 16) ? 16 : 0);
  const _Float16* p = base + (size_t)(rowBase + m) * D + kb;
  return *(const v16h*)(p);
}

// B: 32x16 f16 (K x N). Lane L (0-15): N=L, K=0..15 contiguous;
// lanes 16-31: N=L-16, K=16..31 contiguous.  One 32-byte chunk per lane.
// B[kk][n] = k[colBase+n][kk]  (i.e. k^T, which is what qk^T needs).
__device__ __forceinline__ v16h load_b_frag(const _Float16* __restrict__ base,
                                            int colBase, int kOff, int lane) {
  const int n  = lane & 15;
  const int kb = kOff + ((lane & 16) ? 16 : 0);
  return *(const v16h*)(base + (size_t)(colBase + n) * D + kb);
}

__device__ __forceinline__ v8f wmma_acc(v16h a, v16h b, v8f c) {
  return __builtin_amdgcn_wmma_f32_16x16x32_f16(
      /*neg_a=*/false, a, /*neg_b=*/false, b,
      /*c_mod=*/(short)0, c, /*reuse_a=*/false, /*reuse_b=*/false);
}

// C/D layout: vgpr r, lanes 0-15 -> (M=r, N=lane); lanes 16-31 -> (M=r+8, N=lane-16).

// ---------------------------------------------------------------------------
// Kernel 1: q/k MLPs in fp32, store f16.  4 rows per 256-thread block.
// ---------------------------------------------------------------------------
__global__ void mlp_qk_kernel(const float* __restrict__ x,
                              const float* __restrict__ Wq1, const float* __restrict__ bq1,
                              const float* __restrict__ Wq2, const float* __restrict__ bq2,
                              const float* __restrict__ Wk1, const float* __restrict__ bk1,
                              const float* __restrict__ Wk2, const float* __restrict__ bk2,
                              _Float16* __restrict__ qh, _Float16* __restrict__ kh) {
  __shared__ float xs[4][FIN + 3];
  __shared__ float hs[4][D];

  const int sub = threadIdx.x >> 6;         // row within block (0..3)
  const int d   = threadIdx.x & 63;         // output feature
  const int row = blockIdx.x * 4 + sub;

  for (int f = d; f < FIN; f += 64) xs[sub][f] = x[(size_t)row * FIN + f];
  __syncthreads();

  // ---- q ----
  float acc = bq1[d];
#pragma unroll 4
  for (int f = 0; f < FIN; ++f) acc = fmaf(xs[sub][f], Wq1[f * D + d], acc);
  hs[sub][d] = fmaxf(acc, 0.0f);
  __syncthreads();

  float acc2 = bq2[d];
#pragma unroll 8
  for (int j = 0; j < D; ++j) acc2 = fmaf(hs[sub][j], Wq2[j * D + d], acc2);
  qh[(size_t)row * D + d] = (_Float16)acc2;
  __syncthreads();                           // done reading hs before reuse

  // ---- k ----
  acc = bk1[d];
#pragma unroll 4
  for (int f = 0; f < FIN; ++f) acc = fmaf(xs[sub][f], Wk1[f * D + d], acc);
  hs[sub][d] = fmaxf(acc, 0.0f);
  __syncthreads();

  acc2 = bk2[d];
#pragma unroll 8
  for (int j = 0; j < D; ++j) acc2 = fmaf(hs[sub][j], Wk2[j * D + d], acc2);
  kh[(size_t)row * D + d] = (_Float16)acc2;
}

// ---------------------------------------------------------------------------
// Kernel 2: rowmax[i] = max_j (q_i . k_j).
// One block per 64-row group (RT=4 row tiles held in A-fragments per wave);
// 8 waves stride over 512 column tiles, B loaded once and reused 4x.
// ---------------------------------------------------------------------------
__global__ void rowmax_kernel(const _Float16* __restrict__ qh,
                              const _Float16* __restrict__ kh,
                              float* __restrict__ rowmax) {
  const int rowBase = blockIdx.x * (16 * RT);
  const int wave = threadIdx.x >> 5;
  const int lane = threadIdx.x & 31;

  v16h a0[RT], a1[RT];
#pragma unroll
  for (int t = 0; t < RT; ++t) {
    a0[t] = load_a_frag(qh, rowBase + t * 16, 0,  lane);
    a1[t] = load_a_frag(qh, rowBase + t * 16, 32, lane);
  }

  float mx[RT][8];
#pragma unroll
  for (int t = 0; t < RT; ++t)
#pragma unroll
    for (int r = 0; r < 8; ++r) mx[t][r] = NEG_BIG;

  // 64 iterations; unroll 2 -> two independent WMMA pipelines per iteration
#pragma unroll 2
  for (int jt = wave; jt < NT; jt += 8) {
    const int colBase = jt * 16;
    v16h b0 = load_b_frag(kh, colBase, 0,  lane);
    v16h b1 = load_b_frag(kh, colBase, 32, lane);
    if (jt + 8 < NT)  // prefetch next column tile's k rows (global_prefetch_b8)
      __builtin_prefetch((const void*)(kh + (size_t)((jt + 8) * 16 + (lane & 15)) * D), 0, 1);
#pragma unroll
    for (int t = 0; t < RT; ++t) {
      v8f c = {};
      c = wmma_acc(a0[t], b0, c);
      c = wmma_acc(a1[t], b1, c);
#pragma unroll
      for (int r = 0; r < 8; ++r) mx[t][r] = fmaxf(mx[t][r], c[r]);
    }
  }

  // reduce across the 16 lanes of each half (xor masks stay within half)
#pragma unroll
  for (int off = 1; off < 16; off <<= 1)
#pragma unroll
    for (int t = 0; t < RT; ++t)
#pragma unroll
      for (int r = 0; r < 8; ++r)
        mx[t][r] = fmaxf(mx[t][r], __shfl_xor(mx[t][r], off, 32));

  __shared__ float red[8][16 * RT];
  if ((lane & 15) == 0) {
    const int mAdd = (lane & 16) ? 8 : 0;
#pragma unroll
    for (int t = 0; t < RT; ++t)
#pragma unroll
      for (int r = 0; r < 8; ++r) red[wave][t * 16 + mAdd + r] = mx[t][r];
  }
  __syncthreads();
  if (threadIdx.x < 16 * RT) {
    float m = red[0][threadIdx.x];
#pragma unroll
    for (int w = 1; w < 8; ++w) m = fmaxf(m, red[w][threadIdx.x]);
    rowmax[rowBase + threadIdx.x] = m;
  }
}

// ---------------------------------------------------------------------------
// Kernel 3: online softmax stats per row:  s_ij = TEMP*c_ij / rowmax[j]
//   m_i = max_j s_ij,  l_i = sum_j exp(s_ij - m_i)
// Same RT=4 register blocking as rowmax_kernel.
// ---------------------------------------------------------------------------
__global__ void stats_kernel(const _Float16* __restrict__ qh,
                             const _Float16* __restrict__ kh,
                             const float* __restrict__ rowmax,
                             float* __restrict__ mrow, float* __restrict__ lrow) {
  const int rowBase = blockIdx.x * (16 * RT);
  const int wave = threadIdx.x >> 5;
  const int lane = threadIdx.x & 31;

  v16h a0[RT], a1[RT];
#pragma unroll
  for (int t = 0; t < RT; ++t) {
    a0[t] = load_a_frag(qh, rowBase + t * 16, 0,  lane);
    a1[t] = load_a_frag(qh, rowBase + t * 16, 32, lane);
  }

  float mv[RT][8], lv[RT][8];
#pragma unroll
  for (int t = 0; t < RT; ++t)
#pragma unroll
    for (int r = 0; r < 8; ++r) { mv[t][r] = NEG_BIG; lv[t][r] = 0.0f; }

#pragma unroll 2
  for (int jt = wave; jt < NT; jt += 8) {
    const int colBase = jt * 16;
    const float inv = TEMP / rowmax[colBase + (lane & 15)];  // this lane's column
    v16h b0 = load_b_frag(kh, colBase, 0,  lane);
    v16h b1 = load_b_frag(kh, colBase, 32, lane);
    if (jt + 8 < NT)
      __builtin_prefetch((const void*)(kh + (size_t)((jt + 8) * 16 + (lane & 15)) * D), 0, 1);
#pragma unroll
    for (int t = 0; t < RT; ++t) {
      v8f c = {};
      c = wmma_acc(a0[t], b0, c);
      c = wmma_acc(a1[t], b1, c);
#pragma unroll
      for (int r = 0; r < 8; ++r) {
        float s  = c[r] * inv;
        float nm = fmaxf(mv[t][r], s);
        lv[t][r] = lv[t][r] * __expf(mv[t][r] - nm) + __expf(s - nm);
        mv[t][r] = nm;
      }
    }
  }

  // combine (m,l) across the 16 lanes of each half
#pragma unroll
  for (int off = 1; off < 16; off <<= 1) {
#pragma unroll
    for (int t = 0; t < RT; ++t)
#pragma unroll
      for (int r = 0; r < 8; ++r) {
        float mo = __shfl_xor(mv[t][r], off, 32);
        float lo = __shfl_xor(lv[t][r], off, 32);
        float nm = fmaxf(mv[t][r], mo);
        lv[t][r] = lv[t][r] * __expf(mv[t][r] - nm) + lo * __expf(mo - nm);
        mv[t][r] = nm;
      }
  }

  __shared__ float redm[8][16 * RT];
  __shared__ float redl[8][16 * RT];
  if ((lane & 15) == 0) {
    const int mAdd = (lane & 16) ? 8 : 0;
#pragma unroll
    for (int t = 0; t < RT; ++t)
#pragma unroll
      for (int r = 0; r < 8; ++r) {
        redm[wave][t * 16 + mAdd + r] = mv[t][r];
        redl[wave][t * 16 + mAdd + r] = lv[t][r];
      }
  }
  __syncthreads();
  if (threadIdx.x < 16 * RT) {
    float m = redm[0][threadIdx.x];
    float l = redl[0][threadIdx.x];
#pragma unroll
    for (int w = 1; w < 8; ++w) {
      float mo = redm[w][threadIdx.x];
      float lo = redl[w][threadIdx.x];
      float nm = fmaxf(m, mo);
      l = l * __expf(m - nm) + lo * __expf(mo - nm);
      m = nm;
    }
    mrow[rowBase + threadIdx.x] = m;
    lrow[rowBase + threadIdx.x] = l;
  }
}

// ---------------------------------------------------------------------------
// Kernel 4: out[i][j] = exp(s_ij - m_i) / l_i.
// Block = 512 threads = 16 waves covering 32 rows x 128 cols
// (2 row tiles x 8 col tiles, one 16x16 WMMA tile per wave), staged through
// LDS so the 256 MB output is written as contiguous 512B rows.
// ---------------------------------------------------------------------------
__global__ void out_kernel(const _Float16* __restrict__ qh,
                           const _Float16* __restrict__ kh,
                           const float* __restrict__ rowmax,
                           const float* __restrict__ mrow,
                           const float* __restrict__ lrow,
                           float* __restrict__ out) {
  const int rowBase  = blockIdx.y * 32;
  const int colSuper = blockIdx.x * 128;
  const int wave = threadIdx.x >> 5;
  const int lane = threadIdx.x & 31;
  const int rt   = wave >> 3;                // 0..1  row tile in block
  const int ct   = wave & 7;                 // 0..7  col tile in block
  const int rowB    = rowBase + rt * 16;
  const int colBase = colSuper + ct * 16;

  __shared__ float tile[32][132];            // +4 pad: conflict-free half-wave stores
  __shared__ float sm[32];
  __shared__ float sl[32];
  if (threadIdx.x < 32) {
    sm[threadIdx.x] = mrow[rowBase + threadIdx.x];
    sl[threadIdx.x] = 1.0f / lrow[rowBase + threadIdx.x];
  }

  const float inv = TEMP / rowmax[colBase + (lane & 15)];

  v16h a0 = load_a_frag(qh, rowB, 0,  lane);
  v16h a1 = load_a_frag(qh, rowB, 32, lane);
  v16h b0 = load_b_frag(kh, colBase, 0,  lane);
  v16h b1 = load_b_frag(kh, colBase, 32, lane);
  v8f c = {};
  c = wmma_acc(a0, b0, c);
  c = wmma_acc(a1, b1, c);

  __syncthreads();  // sm/sl ready

  const int mAdd = (lane & 16) ? 8 : 0;
  const int nCol = ct * 16 + (lane & 15);
#pragma unroll
  for (int r = 0; r < 8; ++r) {
    const int M = rt * 16 + r + mAdd;
    float s = c[r] * inv;
    tile[M][nCol] = __expf(s - sm[M]) * sl[M];
  }
  __syncthreads();

  // coalesced writeback: 4096 floats, consecutive threads -> consecutive addrs
#pragma unroll
  for (int i = threadIdx.x; i < 32 * 128; i += 512) {
    const int rr = i >> 7;
    const int cc = i & 127;
    out[(size_t)(rowBase + rr) * N + colSuper + cc] = tile[rr][cc];
  }
}

// ---------------------------------------------------------------------------
extern "C" void kernel_launch(void* const* d_in, const int* in_sizes, int n_in,
                              void* d_out, int out_size, void* d_ws, size_t ws_size,
                              hipStream_t stream) {
  const float* x   = (const float*)d_in[0];
  const float* Wq1 = (const float*)d_in[1];
  const float* bq1 = (const float*)d_in[2];
  const float* Wq2 = (const float*)d_in[3];
  const float* bq2 = (const float*)d_in[4];
  const float* Wk1 = (const float*)d_in[5];
  const float* bk1 = (const float*)d_in[6];
  const float* Wk2 = (const float*)d_in[7];
  const float* bk2 = (const float*)d_in[8];
  float* out = (float*)d_out;

  // workspace: q/k f16 (1 MB each) + rowmax/m/l (32 KB each)
  _Float16* qh = (_Float16*)d_ws;
  _Float16* kh = qh + (size_t)N * D;
  float* rowmax = (float*)(kh + (size_t)N * D);
  float* mrow   = rowmax + N;
  float* lrow   = mrow + N;

  hipLaunchKernelGGL(mlp_qk_kernel, dim3(N / 4), dim3(256), 0, stream,
                     x, Wq1, bq1, Wq2, bq2, Wk1, bk1, Wk2, bk2, qh, kh);
  hipLaunchKernelGGL(rowmax_kernel, dim3(N / (16 * RT)), dim3(256), 0, stream,
                     qh, kh, rowmax);
  hipLaunchKernelGGL(stats_kernel, dim3(N / (16 * RT)), dim3(256), 0, stream,
                     qh, kh, rowmax, mrow, lrow);
  hipLaunchKernelGGL(out_kernel, dim3(N / 128, N / 32), dim3(512), 0, stream,
                     qh, kh, rowmax, mrow, lrow, out);
}
